// PartialEncoderWeightedSumEDDIMultiWeightATSE_6846177870202
// MI455X (gfx1250) — compile-verified, matrix operand order checked by reading, not run on hardware
//
#include <hip/hip_runtime.h>
#include <hip/hip_bf16.h>
#include <math.h>

typedef _Float16 half_t;
typedef __attribute__((ext_vector_type(16))) _Float16 v16h;
typedef __attribute__((ext_vector_type(8)))  _Float16 v8h;
typedef __attribute__((ext_vector_type(8)))  float    v8f;

static constexpr int cB  = 64;
static constexpr int cJ  = 16384;
static constexpr int cD  = 32;
static constexpr int cL  = 32;
static constexpr int cP  = 8;            // partials per batch in pass 2
static constexpr int cJP = cJ / cP;      // junctions per partial (2048)

// K-index map for 16-bit WMMA A/B fragments (16x32 K-tile, wave32):
// element i (0..15); v=i/2, pos=i%2, hs=lane>>4
// k = ((v<4)?0:16) + 8*hs + (v&3)*2 + pos
// => per lane, elements 0..7 cover k = hs*8 + 0..7 (contiguous),
//    elements 8..15 cover k = 16 + hs*8 + 0..7 (contiguous).
__device__ __forceinline__ int frag_k(int v, int pos, int hs) {
    return ((v < 4) ? 0 : 16) + hs * 8 + (v & 3) * 2 + pos;
}

// Stage a row-major f32 weight (Ksrc x Nsrc) into LDS as f16 WMMA B-fragments,
// fragment-major: dst[f*512 + lane*16 + elem], f = ks*nTilesN + t.
__device__ __forceinline__ void stage_frags(const float* __restrict__ W, half_t* dst,
                                            int nTilesN, int nFrags, int Ksrc, int Nsrc,
                                            int tid) {
    for (int idx = tid; idx < nFrags * 512; idx += 256) {
        int f = idx >> 9;
        int within = idx & 511;
        int lane = within >> 4;
        int e = within & 15;
        int ks = f / nTilesN, t = f - ks * nTilesN;
        int hs2 = lane >> 4;
        int n = t * 16 + (lane & 15);
        int v = e >> 1, pos = e & 1;
        int k = ks * 32 + frag_k(v, pos, hs2);
        float val = (k < Ksrc && n < Nsrc) ? W[k * Nsrc + n] : 0.f;
        dst[idx] = (half_t)val;
    }
}

__device__ __forceinline__ v16h load_frag16(const half_t* region, int f, int lane) {
    return *(const v16h*)(region + f * 512 + lane * 16);
}

// ---------------------------------------------------------------------------
// Pass 1: per-junction h-MLP + gate MLP, all three GEMMs on WMMA.
// grid = B * (J/128), block = 256 (8 waves, 16 junctions per wave)
// ---------------------------------------------------------------------------
__global__ __launch_bounds__(256) void pass1_kernel(
    const float* __restrict__ x, const int* __restrict__ mask,
    const int* __restrict__ aidx, const float* __restrict__ fe,
    const float* __restrict__ aemb,
    const float* __restrict__ hW1, const float* __restrict__ hb1,
    const float* __restrict__ hg1, const float* __restrict__ hbt1,
    const float* __restrict__ hW2, const float* __restrict__ hb2,
    const float* __restrict__ hg2, const float* __restrict__ hbt2,
    const float* __restrict__ gW1, const float* __restrict__ gb1,
    const float* __restrict__ gW2, const float* __restrict__ gb2,
    float* __restrict__ logits, half_t* __restrict__ hout)
{
    __shared__ alignas(32) half_t sW1f[16 * 512];  // hW1 frags: 2 ks x 8 ntiles
    __shared__ alignas(32) half_t sW2f[8 * 512];   // hW2 frags: 4 ks x 2 ntiles
    __shared__ alignas(32) half_t sGWf[4 * 512];   // gW1 frags: 2 ks x 2 ntiles
    __shared__ alignas(32) half_t aA2[8][4 * 512]; // per-wave A frags for GEMM2 (K=128)
    __shared__ alignas(32) half_t aAg[8][2 * 512]; // per-wave A frags for gate GEMM (K=64)
    __shared__ alignas(16) float  scr[8][16 * 128];
    __shared__ alignas(16) float  sb1[128], sg1[128], sB1[128];
    __shared__ float  sb2[32], sg2[32], sB2[32];
    __shared__ float  sgb1p[32], sgW2[24 * 4], sgb2[4];

    const int tid = threadIdx.x;

    stage_frags(hW1, sW1f, 8, 16, 33, 128, tid);
    stage_frags(hW2, sW2f, 2, 8, 128, 32, tid);
    stage_frags(gW1, sGWf, 2, 4, 48, 24, tid);
    if (tid < 128) { sb1[tid] = hb1[tid]; sg1[tid] = hg1[tid]; sB1[tid] = hbt1[tid]; }
    if (tid < 32)  { sb2[tid] = hb2[tid]; sg2[tid] = hg2[tid]; sB2[tid] = hbt2[tid]; }
    if (tid < 32)  sgb1p[tid] = (tid < 24) ? gb1[tid] : 0.f;
    if (tid < 96)  sgW2[tid] = gW2[tid];
    if (tid < 4)   sgb2[tid] = gb2[tid];
    __syncthreads();

    const int wave = tid >> 5, lane = tid & 31;
    const int bb = blockIdx.x >> 7;
    const int jb = blockIdx.x & 127;
    const int j0 = jb * 128 + wave * 16;
    const int m = lane & 15, hs = lane >> 4;
    const int j = j0 + m;

    float*  myscr = scr[wave];
    half_t* aA2w  = aA2[wave];
    half_t* aAgw  = aAg[wave];

    // ================= GEMM1: h_in(16x64pad) x hW1(64x128) =================
    v8f acc[8];
#pragma unroll
    for (int t = 0; t < 8; ++t)
#pragma unroll
        for (int e = 0; e < 8; ++e) acc[t][e] = 0.f;

    const float* ferow = fe + (size_t)j * 32;
    const float  xv    = x[(size_t)bb * cJ + j];

#pragma unroll
    for (int ks = 0; ks < 2; ++ks) {
        v16h a;
#pragma unroll
        for (int i = 0; i < 16; ++i) {
            int v = i >> 1, pos = i & 1;
            int k = ks * 32 + frag_k(v, pos, hs);
            float val = (k == 0) ? xv : ((k <= 32) ? ferow[k - 1] : 0.f);
            a[i] = (half_t)val;
        }
#pragma unroll
        for (int t = 0; t < 8; ++t) {
            v16h bf = load_frag16(sW1f, ks * 8 + t, lane);
            acc[t] = __builtin_amdgcn_wmma_f32_16x16x32_f16(
                false, a, false, bf, (short)0, acc[t], false, false);
        }
    }

#pragma unroll
    for (int t = 0; t < 8; ++t)
#pragma unroll
        for (int r = 0; r < 8; ++r) {
            int mm = r + hs * 8, nn = t * 16 + m;
            myscr[mm * 128 + nn] = acc[t][r] + sb1[nn];
        }
    __syncthreads();

    // LN over H=128 + ReLU; emit directly as packed GEMM2 A-fragments.
    if (lane < 16) {
        const float4* row4 = (const float4*)(myscr + m * 128);
        float s = 0.f, s2 = 0.f;
#pragma unroll
        for (int q = 0; q < 32; ++q) {
            float4 vv = row4[q];
            s  += vv.x + vv.y + vv.z + vv.w;
            s2 += vv.x * vv.x + vv.y * vv.y + vv.z * vv.z + vv.w * vv.w;
        }
        float mean = s * (1.f / 128.f);
        float var  = s2 * (1.f / 128.f) - mean * mean;
        float inv  = 1.f / sqrtf(var + 1e-5f);
        const float4* g4 = (const float4*)sg1;
        const float4* b4 = (const float4*)sB1;
#pragma unroll
        for (int g = 0; g < 16; ++g) {   // 8 consecutive k per group
            float4 a0 = row4[g * 2], a1 = row4[g * 2 + 1];
            float4 gg0 = g4[g * 2], gg1 = g4[g * 2 + 1];
            float4 bb0 = b4[g * 2], bb1 = b4[g * 2 + 1];
            v8h p;
            p[0] = (half_t)fmaxf((a0.x - mean) * inv * gg0.x + bb0.x, 0.f);
            p[1] = (half_t)fmaxf((a0.y - mean) * inv * gg0.y + bb0.y, 0.f);
            p[2] = (half_t)fmaxf((a0.z - mean) * inv * gg0.z + bb0.z, 0.f);
            p[3] = (half_t)fmaxf((a0.w - mean) * inv * gg0.w + bb0.w, 0.f);
            p[4] = (half_t)fmaxf((a1.x - mean) * inv * gg1.x + bb1.x, 0.f);
            p[5] = (half_t)fmaxf((a1.y - mean) * inv * gg1.y + bb1.y, 0.f);
            p[6] = (half_t)fmaxf((a1.z - mean) * inv * gg1.z + bb1.z, 0.f);
            p[7] = (half_t)fmaxf((a1.w - mean) * inv * gg1.w + bb1.w, 0.f);
            int ks = g >> 2, r32 = g & 3;
            int tl = m + ((r32 & 1) << 4);
            int off8 = (r32 & 2) ? 8 : 0;
            *(v8h*)(aA2w + ks * 512 + tl * 16 + off8) = p;
        }
    }
    __syncthreads();

    // ================= GEMM2: h1(16x128) x hW2(128x32) =================
    v8f acc2[2];
#pragma unroll
    for (int t = 0; t < 2; ++t)
#pragma unroll
        for (int e = 0; e < 8; ++e) acc2[t][e] = 0.f;

#pragma unroll
    for (int ks = 0; ks < 4; ++ks) {
        v16h a = *(const v16h*)(aA2w + ks * 512 + lane * 16);
#pragma unroll
        for (int t = 0; t < 2; ++t) {
            v16h bf = load_frag16(sW2f, ks * 2 + t, lane);
            acc2[t] = __builtin_amdgcn_wmma_f32_16x16x32_f16(
                false, a, false, bf, (short)0, acc2[t], false, false);
        }
    }
#pragma unroll
    for (int t = 0; t < 2; ++t)
#pragma unroll
        for (int r = 0; r < 8; ++r) {
            int mm = r + hs * 8, nn = t * 16 + m;
            myscr[mm * 32 + nn] = acc2[t][r] + sb2[nn];
        }
    __syncthreads();

    // LN over D=32 + ReLU; persist h_out (f16) and build gate A-fragments.
    if (lane < 16) {
        const float4* row4 = (const float4*)(myscr + m * 32);
        float4 rv[8];
#pragma unroll
        for (int q = 0; q < 8; ++q) rv[q] = row4[q];
        float s = 0.f, s2 = 0.f;
#pragma unroll
        for (int q = 0; q < 8; ++q) {
            s  += rv[q].x + rv[q].y + rv[q].z + rv[q].w;
            s2 += rv[q].x * rv[q].x + rv[q].y * rv[q].y + rv[q].z * rv[q].z + rv[q].w * rv[q].w;
        }
        float mean = s * (1.f / 32.f);
        float var  = s2 * (1.f / 32.f) - mean * mean;
        float inv  = 1.f / sqrtf(var + 1e-5f);
        float hv[32];
#pragma unroll
        for (int q = 0; q < 8; ++q) {
            hv[q * 4 + 0] = fmaxf((rv[q].x - mean) * inv * sg2[q * 4 + 0] + sB2[q * 4 + 0], 0.f);
            hv[q * 4 + 1] = fmaxf((rv[q].y - mean) * inv * sg2[q * 4 + 1] + sB2[q * 4 + 1], 0.f);
            hv[q * 4 + 2] = fmaxf((rv[q].z - mean) * inv * sg2[q * 4 + 2] + sB2[q * 4 + 2], 0.f);
            hv[q * 4 + 3] = fmaxf((rv[q].w - mean) * inv * sg2[q * 4 + 3] + sB2[q * 4 + 3], 0.f);
        }
        const size_t hbase = ((size_t)bb * cJ + j) * 32;
#pragma unroll
        for (int q = 0; q < 4; ++q) {
            v8h p;
#pragma unroll
            for (int i = 0; i < 8; ++i) p[i] = (half_t)hv[q * 8 + i];
            *(v8h*)(hout + hbase + q * 8) = p;
        }
#pragma unroll
        for (int hs2 = 0; hs2 < 2; ++hs2) {
#pragma unroll
            for (int run = 0; run < 2; ++run) {
                v8h p;
#pragma unroll
                for (int i = 0; i < 8; ++i) p[i] = (half_t)hv[run * 16 + hs2 * 8 + i];
                *(v8h*)(aAgw + (m + hs2 * 16) * 16 + run * 8) = p;
            }
        }
        const float4* ae4 = (const float4*)(aemb + (size_t)aidx[j] * 16);
        float4 a0 = ae4[0], a1 = ae4[1], a2 = ae4[2], a3 = ae4[3];
        float aev[16] = { a0.x, a0.y, a0.z, a0.w, a1.x, a1.y, a1.z, a1.w,
                          a2.x, a2.y, a2.z, a2.w, a3.x, a3.y, a3.z, a3.w };
#pragma unroll
        for (int hs2 = 0; hs2 < 2; ++hs2) {
            v8h p, z;
#pragma unroll
            for (int i = 0; i < 8; ++i) { p[i] = (half_t)aev[hs2 * 8 + i]; z[i] = (half_t)0.f; }
            *(v8h*)(aAgw + 512 + (m + hs2 * 16) * 16 + 0) = p;
            *(v8h*)(aAgw + 512 + (m + hs2 * 16) * 16 + 8) = z;
        }
    }
    __syncthreads();

    // ================= gate GEMM: gate_in(16x64) x gW1(64x32pad) =================
    v8f accg[2];
#pragma unroll
    for (int t = 0; t < 2; ++t)
#pragma unroll
        for (int e = 0; e < 8; ++e) accg[t][e] = 0.f;

#pragma unroll
    for (int ks = 0; ks < 2; ++ks) {
        v16h a = *(const v16h*)(aAgw + ks * 512 + lane * 16);
#pragma unroll
        for (int t = 0; t < 2; ++t) {
            v16h bf = load_frag16(sGWf, ks * 2 + t, lane);
            accg[t] = __builtin_amdgcn_wmma_f32_16x16x32_f16(
                false, a, false, bf, (short)0, accg[t], false, false);
        }
    }
#pragma unroll
    for (int t = 0; t < 2; ++t)
#pragma unroll
        for (int r = 0; r < 8; ++r) {
            int mm = r + hs * 8, nn = t * 16 + m;
            myscr[512 + mm * 32 + nn] = accg[t][r] + sgb1p[nn];
        }
    __syncthreads();

    // gate layer 2 (24 -> 4), clip, mask, emit logits
    if (lane < 16) {
        const float* grow = myscr + 512 + m * 32;
        float raw[4] = { sgb2[0], sgb2[1], sgb2[2], sgb2[3] };
#pragma unroll
        for (int q = 0; q < 24; ++q) {
            float r = fmaxf(grow[q], 0.f);
#pragma unroll
            for (int w = 0; w < 4; ++w) raw[w] += r * sgW2[q * 4 + w];
        }
        const int mk = mask[(size_t)bb * cJ + j];
        float4 lg;
        lg.x = mk ? fminf(fmaxf(raw[0], -10.f), 10.f) : -1e9f;
        lg.y = mk ? fminf(fmaxf(raw[1], -10.f), 10.f) : -1e9f;
        lg.z = mk ? fminf(fmaxf(raw[2], -10.f), 10.f) : -1e9f;
        lg.w = mk ? fminf(fmaxf(raw[3], -10.f), 10.f) : -1e9f;
        *(float4*)(logits + ((size_t)bb * cJ + j) * 4) = lg;
    }
}

// ---------------------------------------------------------------------------
// Pass 2a: partial online-softmax reduction. grid = B*P, block = 256.
// Each block handles J/P junctions of one batch: local max, local sumexp,
// local weighted-sum vector (numerator), using local max as reference.
// ---------------------------------------------------------------------------
__global__ __launch_bounds__(256) void pass2a_kernel(
    const float* __restrict__ logits, const half_t* __restrict__ hout,
    float* __restrict__ pvec, float* __restrict__ psum, float* __restrict__ pmax)
{
    __shared__ float  red[256];
    __shared__ float  bmax[4], ssum[4];
    __shared__ alignas(16) half_t lh[256 * 32];
    __shared__ float  lp[256 * 4];

    const int tid = threadIdx.x;
    const int b = blockIdx.x >> 3, part = blockIdx.x & 7;
    const int jstart = part * cJP;
    const float4* lg4 = (const float4*)(logits + (size_t)b * cJ * 4);

    // local max over assigned range
    float mx[4] = { -3e38f, -3e38f, -3e38f, -3e38f };
    for (int c = 0; c < cJP / 256; ++c) {
        float4 l = lg4[jstart + c * 256 + tid];
        mx[0] = fmaxf(mx[0], l.x); mx[1] = fmaxf(mx[1], l.y);
        mx[2] = fmaxf(mx[2], l.z); mx[3] = fmaxf(mx[3], l.w);
    }
    for (int w = 0; w < 4; ++w) {
        red[tid] = mx[w]; __syncthreads();
        for (int s = 128; s > 0; s >>= 1) {
            if (tid < s) red[tid] = fmaxf(red[tid], red[tid + s]);
            __syncthreads();
        }
        if (tid == 0) bmax[w] = red[0];
        __syncthreads();
    }

    // exp + partial weighted sums; 2 threads per (w,d): halves of each chunk
    float se[4] = { 0.f, 0.f, 0.f, 0.f };
    const int p2 = tid >> 7, w = (tid >> 5) & 3, d = tid & 31;
    float acc = 0.f;
    const uint4* hsrc = (const uint4*)(hout + (size_t)b * cJ * 32);
    uint4* dsth = (uint4*)lh;

    for (int c = 0; c < cJP / 256; ++c) {
        const int jj0 = jstart + c * 256;
        const uint4* srow = hsrc + (size_t)(jj0 + tid) * 4;
        dsth[tid * 4 + 0] = srow[0]; dsth[tid * 4 + 1] = srow[1];
        dsth[tid * 4 + 2] = srow[2]; dsth[tid * 4 + 3] = srow[3];
        float4 l = lg4[jj0 + tid];
        float p0 = __expf(l.x - bmax[0]); float p1 = __expf(l.y - bmax[1]);
        float p2e = __expf(l.z - bmax[2]); float p3 = __expf(l.w - bmax[3]);
        lp[tid * 4 + 0] = p0; lp[tid * 4 + 1] = p1;
        lp[tid * 4 + 2] = p2e; lp[tid * 4 + 3] = p3;
        se[0] += p0; se[1] += p1; se[2] += p2e; se[3] += p3;
        __syncthreads();
        const int q0 = p2 * 128;
        for (int q = q0; q < q0 + 128; ++q)
            acc += lp[q * 4 + w] * (float)lh[q * 32 + d];
        __syncthreads();
    }

    for (int ww = 0; ww < 4; ++ww) {
        red[tid] = se[ww]; __syncthreads();
        for (int s = 128; s > 0; s >>= 1) {
            if (tid < s) red[tid] += red[tid + s];
            __syncthreads();
        }
        if (tid == 0) ssum[ww] = red[0];
        __syncthreads();
    }

    red[tid] = acc; __syncthreads();
    const int pid = b * cP + part;
    if (tid < 128) pvec[(size_t)pid * 128 + tid] = red[tid] + red[tid + 128];
    if (tid < 4)  { psum[pid * 4 + tid] = ssum[tid]; pmax[pid * 4 + tid] = bmax[tid]; }
}

// ---------------------------------------------------------------------------
// Pass 2b: merge P partials per batch (online-softmax merge) -> head, flags.
// ---------------------------------------------------------------------------
__global__ __launch_bounds__(256) void pass2b_kernel(
    const float* __restrict__ pvec, const float* __restrict__ psum,
    const float* __restrict__ pmax, float* __restrict__ head, int* __restrict__ flags)
{
    __shared__ float M[4], denom[4], scal[cP * 4];
    const int tid = threadIdx.x, b = blockIdx.x;

    if (tid < 4) {
        float mm = -3e38f;
        for (int i = 0; i < cP; ++i) mm = fmaxf(mm, pmax[(b * cP + i) * 4 + tid]);
        M[tid] = mm;
    }
    __syncthreads();
    if (tid < cP * 4) {
        int i = tid >> 2, ww = tid & 3;
        scal[tid] = __expf(pmax[(b * cP + i) * 4 + ww] - M[ww]);
    }
    __syncthreads();
    if (tid < 4) {
        float s = 0.f;
        for (int i = 0; i < cP; ++i) s += psum[(b * cP + i) * 4 + tid] * scal[i * 4 + tid];
        denom[tid] = s;
    }
    __syncthreads();
    if (tid < 128) {
        const int w = tid >> 5;
        float s = 0.f;
        for (int i = 0; i < cP; ++i)
            s += pvec[(size_t)(b * cP + i) * 128 + tid] * scal[i * 4 + w];
        head[(size_t)b * 128 + tid] = s / denom[w];
    }
    if (tid == 0) flags[b] = (M[0] > -5e8f) ? 1 : 0;
}

// ---------------------------------------------------------------------------
// Pass 3: per-batch combiner + encoder -> mu, logvar
// ---------------------------------------------------------------------------
__global__ __launch_bounds__(256) void pass3_kernel(
    const float* __restrict__ head, const int* __restrict__ flags,
    const float* __restrict__ cWp, const float* __restrict__ cbp,
    const float* __restrict__ cgp, const float* __restrict__ cbtp,
    const float* __restrict__ eW1, const float* __restrict__ eb1,
    const float* __restrict__ eg1, const float* __restrict__ ebt1,
    const float* __restrict__ eW2, const float* __restrict__ eb2,
    const float* __restrict__ eg2, const float* __restrict__ ebt2,
    float* __restrict__ out)
{
    __shared__ float hrow[128], craw[32], cact[32], eact[256], mlr[64], red[256];
    const int tid = threadIdx.x, b = blockIdx.x;

    if (tid < 128) hrow[tid] = head[(size_t)b * 128 + tid];
    __syncthreads();

    if (tid < 32) {
        float s = cbp[tid];
        for (int k = 0; k < 128; ++k) s += hrow[k] * cWp[k * 32 + tid];
        craw[tid] = s;
    }
    __syncthreads();
    if (tid < 32) {
        float su = 0.f, s2 = 0.f;
        for (int k = 0; k < 32; ++k) { float v = craw[k]; su += v; s2 += v * v; }
        float mean = su * (1.f / 32.f);
        float var  = s2 * (1.f / 32.f) - mean * mean;
        float inv  = 1.f / sqrtf(var + 1e-5f);
        float v = (craw[tid] - mean) * inv * cgp[tid] + cbtp[tid];
        v = v > 0.f ? v : 0.f;
        if (!flags[b]) v = 0.f;
        cact[tid] = v;
    }
    __syncthreads();

    float es = eb1[tid];
    for (int k = 0; k < 32; ++k) es += cact[k] * eW1[k * 256 + tid];
    red[tid] = es; __syncthreads();
    for (int s = 128; s > 0; s >>= 1) {
        if (tid < s) red[tid] += red[tid + s];
        __syncthreads();
    }
    float mean = red[0] * (1.f / 256.f);
    __syncthreads();
    float dv = es - mean;
    red[tid] = dv * dv; __syncthreads();
    for (int s = 128; s > 0; s >>= 1) {
        if (tid < s) red[tid] += red[tid + s];
        __syncthreads();
    }
    float var = red[0] * (1.f / 256.f);
    __syncthreads();
    {
        float inv = 1.f / sqrtf(var + 1e-5f);
        float v = dv * inv * eg1[tid] + ebt1[tid];
        eact[tid] = v > 0.f ? v : 0.f;
    }
    __syncthreads();

    if (tid < 64) {
        float s = eb2[tid];
        for (int k = 0; k < 256; ++k) s += eact[k] * eW2[k * 64 + tid];
        mlr[tid] = s;
    }
    __syncthreads();
    if (tid < 64) {
        float su = 0.f, s2 = 0.f;
        for (int k = 0; k < 64; ++k) { float v = mlr[k]; su += v; s2 += v * v; }
        float mean2 = su * (1.f / 64.f);
        float var2  = s2 * (1.f / 64.f) - mean2 * mean2;
        float inv2  = 1.f / sqrtf(var2 + 1e-5f);
        float v = (mlr[tid] - mean2) * inv2 * eg2[tid] + ebt2[tid];
        v = v > 0.f ? v : 0.f;
        if (tid < 32) out[(size_t)b * 32 + tid] = v;
        else          out[(size_t)cB * cL + (size_t)b * 32 + (tid - 32)] = v;
    }
}

// ---------------------------------------------------------------------------
extern "C" void kernel_launch(void* const* d_in, const int* in_sizes, int n_in,
                              void* d_out, int out_size, void* d_ws, size_t ws_size,
                              hipStream_t stream) {
    const float* x    = (const float*)d_in[0];
    const int*   mask = (const int*)d_in[1];
    const int*   aidx = (const int*)d_in[2];
    const float* fe   = (const float*)d_in[3];
    const float* aemb = (const float*)d_in[4];
    const float* hW1  = (const float*)d_in[5];
    const float* hb1  = (const float*)d_in[6];
    const float* hg1  = (const float*)d_in[7];
    const float* hbt1 = (const float*)d_in[8];
    const float* hW2  = (const float*)d_in[9];
    const float* hb2  = (const float*)d_in[10];
    const float* hg2  = (const float*)d_in[11];
    const float* hbt2 = (const float*)d_in[12];
    const float* gW1  = (const float*)d_in[13];
    const float* gb1  = (const float*)d_in[14];
    const float* gW2  = (const float*)d_in[15];
    const float* gb2  = (const float*)d_in[16];
    const float* cWp  = (const float*)d_in[17];
    const float* cbp  = (const float*)d_in[18];
    const float* cgp  = (const float*)d_in[19];
    const float* cbtp = (const float*)d_in[20];
    const float* eW1  = (const float*)d_in[21];
    const float* eb1  = (const float*)d_in[22];
    const float* eg1  = (const float*)d_in[23];
    const float* ebt1 = (const float*)d_in[24];
    const float* eW2  = (const float*)d_in[25];
    const float* eb2  = (const float*)d_in[26];
    const float* eg2  = (const float*)d_in[27];
    const float* ebt2 = (const float*)d_in[28];

    char* ws = (char*)d_ws;
    const size_t off_logits = 0;
    const size_t off_hout   = off_logits + (size_t)cB * cJ * 4 * sizeof(float);
    const size_t off_head   = off_hout   + (size_t)cB * cJ * cD * sizeof(half_t);
    const size_t off_flags  = off_head   + (size_t)cB * 128 * sizeof(float);
    const size_t off_pvec   = off_flags  + (size_t)cB * sizeof(int) + 64;
    const size_t off_psum   = off_pvec   + (size_t)cB * cP * 128 * sizeof(float);
    const size_t off_pmax   = off_psum   + (size_t)cB * cP * 4 * sizeof(float);

    float*  logits = (float*)(ws + off_logits);
    half_t* hout   = (half_t*)(ws + off_hout);
    float*  headp  = (float*)(ws + off_head);
    int*    flags  = (int*)(ws + off_flags);
    float*  pvec   = (float*)(ws + off_pvec);
    float*  psum   = (float*)(ws + off_psum);
    float*  pmax   = (float*)(ws + off_pmax);
    float*  out    = (float*)d_out;

    pass1_kernel<<<dim3(cB * (cJ / 128)), dim3(256), 0, stream>>>(
        x, mask, aidx, fe, aemb,
        hW1, hb1, hg1, hbt1, hW2, hb2, hg2, hbt2,
        gW1, gb1, gW2, gb2, logits, hout);

    pass2a_kernel<<<dim3(cB * cP), dim3(256), 0, stream>>>(logits, hout, pvec, psum, pmax);

    pass2b_kernel<<<dim3(cB), dim3(256), 0, stream>>>(pvec, psum, pmax, headp, flags);

    pass3_kernel<<<dim3(cB), dim3(256), 0, stream>>>(
        headp, flags, cWp, cbp, cgp, cbtp,
        eW1, eb1, eg1, ebt1, eW2, eb2, eg2, ebt2, out);
}
